// Transolver_block_21191368638823
// MI455X (gfx1250) — matrix operand order repaired and checked by reading, not compiled, not run on hardware
//
#include <hip/hip_runtime.h>
#include <hip/hip_bf16.h>

// ---------------------------------------------------------------------------
// Types for CDNA5 WMMA (gfx1250, wave32)
// ---------------------------------------------------------------------------
typedef __attribute__((ext_vector_type(16))) __bf16 bf16x16;
typedef __attribute__((ext_vector_type(8)))  __bf16 bf16x8;
typedef __attribute__((ext_vector_type(8)))  float  v8f;

__device__ __forceinline__ __bf16 f2bf(float f) {
    union { float f; unsigned u; } v; v.f = f;
    unsigned r = v.u + 0x7FFFu + ((v.u >> 16) & 1u);   // round-to-nearest-even
    unsigned short h = (unsigned short)(r >> 16);
    return __builtin_bit_cast(__bf16, h);
}
__device__ __forceinline__ float bf2f(__bf16 b) {
    unsigned short h = __builtin_bit_cast(unsigned short, b);
    union { unsigned u; float f; } v; v.u = ((unsigned)h) << 16;
    return v.f;
}

#define NTOK   32768
#define BB     4
#define HH     8
#define DD     32
#define GG     64
#define CC     256
#define MROWS  (BB * NTOK)     // 131072

// ---------------------------------------------------------------------------
// Weight convert + transpose: Wt[n*K + k] = (bf16) W[k*N + n]
// ---------------------------------------------------------------------------
__global__ void convert_transpose(const float* __restrict__ W,
                                  __bf16* __restrict__ Wt, int K, int N) {
    long long idx = (long long)blockIdx.x * 256 + threadIdx.x;
    if (idx >= (long long)K * N) return;
    int k = (int)(idx / N), n = (int)(idx % N);
    Wt[(long long)n * K + k] = f2bf(W[idx]);
}

// ---------------------------------------------------------------------------
// Row LayerNorm (C=256) -> bf16
// ---------------------------------------------------------------------------
__global__ __launch_bounds__(256) void layernorm_to_bf16(
    const float* __restrict__ X, const float* __restrict__ g,
    const float* __restrict__ b, __bf16* __restrict__ Y) {
    __shared__ float red[256];
    const int t = threadIdx.x;
    const long long row = blockIdx.x;
    float x = X[row * CC + t];
    red[t] = x; __syncthreads();
    for (int s = 128; s > 0; s >>= 1) { if (t < s) red[t] += red[t + s]; __syncthreads(); }
    float mean = red[0] * (1.0f / CC); __syncthreads();
    float d = x - mean;
    red[t] = d * d; __syncthreads();
    for (int s = 128; s > 0; s >>= 1) { if (t < s) red[t] += red[t + s]; __syncthreads(); }
    float var = red[0] * (1.0f / CC);
    float y = d * rsqrtf(var + 1e-5f) * g[t] + b[t];
    Y[row * CC + t] = f2bf(y);
}

// ---------------------------------------------------------------------------
// Generic bf16 WMMA GEMM: C[M, NSUB*16] tile = A[M,KTOT] * Bt[ncols,KTOT]^T.
// Block: 256 threads = 8 waves. Tile 128 (M) x NSUB*16 (N); each wave owns
// 16 rows x NSUB WMMA 16x16 subtiles; K fully unrolled in steps of 32 via
// v_wmma_f32_16x16x32_bf16 (branch-free hot loop, vector b128 loads).
//
// A fragment: row = lane&15; K in [half*16, half*16+16).
// B fragment: col = lane&15; K in [half*16, half*16+16).
// D: VGPR vi -> row vi + half*8, col = lane&15.
// ---------------------------------------------------------------------------
#define EPI_HEADMAJOR 0   // bf16 store remapped (b,n)x(h,d) -> (b,h,n,d)
#define EPI_PLAIN     1   // bf16 store with batched (b,h) column offset
#define EPI_RESID     2   // f32: val + bias + resid  -> Cout
#define EPI_GELU      3   // bf16: gelu(val + bias)   -> Cout
#define EPI_ACC       4   // f32: Cout += val (+bias)

template <int EPI, int NSUB, int KTOT>
__global__ __launch_bounds__(256) void gemm_wmma_bf16(
    const __bf16* __restrict__ A, long long lda,
    const __bf16* __restrict__ Bt, long long ldb,
    const float* __restrict__ bias,
    void* __restrict__ Cout, const float* __restrict__ resid,
    long long strideAz, long long strideBz,
    long long strideCb, long long strideCh, int ldc) {
    const int tid  = threadIdx.x;
    const int wave = tid >> 5;
    const int lane = tid & 31;
    const int half = lane >> 4;
    const int l15  = lane & 15;

    const int row0 = blockIdx.y * 128 + wave * 16;
    const int col0 = blockIdx.x * (NSUB * 16);
    const int z    = blockIdx.z;

    const __bf16* Az = A  + (long long)z * strideAz;
    const __bf16* Bz = Bt + (long long)z * strideBz;
    const long long cOffZ =
        (long long)(z >> 3) * strideCb + (long long)(z & 7) * strideCh;

    v8f acc[NSUB] = {};
    const __bf16* pa_row = Az + (long long)(row0 + l15) * lda + half * 16;
    const __bf16* pb_col = Bz + (long long)(col0 + l15) * ldb + half * 16;

#pragma unroll
    for (int k0 = 0; k0 < KTOT; k0 += 32) {
        // A fragment: two aligned b128 loads, contiguous K (matches B)
        bf16x8 alo = *(const bf16x8*)(pa_row + k0);
        bf16x8 ahi = *(const bf16x8*)(pa_row + k0 + 8);
        bf16x16 afrag = __builtin_shufflevector(
            alo, ahi, 0, 1, 2, 3, 4, 5, 6, 7, 8, 9, 10, 11, 12, 13, 14, 15);
        // B fragments: 32 contiguous aligned bytes per subtile
        bf16x16 bfrag[NSUB];
#pragma unroll
        for (int s = 0; s < NSUB; ++s)
            bfrag[s] = *(const bf16x16*)(pb_col + (long long)s * 16 * ldb + k0);
#pragma unroll
        for (int s = 0; s < NSUB; ++s)
            acc[s] = __builtin_amdgcn_wmma_f32_16x16x32_bf16(
                false, afrag, false, bfrag[s], (short)0, acc[s], false, false);
    }

#pragma unroll
    for (int s = 0; s < NSUB; ++s) {
        int c = col0 + s * 16 + l15;
        float bval = bias ? bias[c] : 0.0f;
#pragma unroll
        for (int vi = 0; vi < 8; ++vi) {
            int r = row0 + vi + half * 8;
            float val = acc[s][vi] + bval;
            if (EPI == EPI_HEADMAJOR) {
                int bidx = r >> 15, n = r & (NTOK - 1);
                int h = c >> 5, d = c & 31;
                long long off =
                    ((long long)(bidx * HH + h) * NTOK + n) * DD + d;
                ((__bf16*)Cout)[off] = f2bf(val);
            } else if (EPI == EPI_PLAIN) {
                ((__bf16*)Cout)[cOffZ + (long long)r * ldc + c] = f2bf(val);
            } else if (EPI == EPI_RESID) {
                long long idx = (long long)r * ldc + c;
                ((float*)Cout)[idx] = val + resid[idx];
            } else if (EPI == EPI_GELU) {
                float ge = 0.5f * val * (1.0f + erff(val * 0.70710678118654752f));
                ((__bf16*)Cout)[(long long)r * ldc + c] = f2bf(ge);
            } else { // EPI_ACC
                long long idx = (long long)r * ldc + c;
                ((float*)Cout)[idx] += val;
            }
        }
    }
}

// ---------------------------------------------------------------------------
// Slice softmax: logits = (x_mid @ w_slice + b_slice)/temp[h]; softmax over G.
// x_mid bf16 (b,h,n,d); sw bf16 (b,h,n,g). 4 tokens per 256-thread block.
// ---------------------------------------------------------------------------
__global__ __launch_bounds__(256) void slice_softmax(
    const __bf16* __restrict__ xm, const float* __restrict__ wsl,
    const float* __restrict__ bsl, const float* __restrict__ temp,
    __bf16* __restrict__ sw) {
    __shared__ float ws[DD * GG];
    __shared__ float xrow[4][DD];
    __shared__ float red[4][GG];
    const int t = threadIdx.x;
    for (int i = t; i < DD * GG; i += 256) ws[i] = wsl[i];
    const int sub = t >> 6;
    const int g   = t & 63;
    const long long tok = (long long)blockIdx.x * 4 + sub;
    if (g < DD) xrow[sub][g] = bf2f(xm[tok * DD + g]);
    __syncthreads();
    float invt = 1.0f / temp[(int)((tok >> 15) & 7)];
    float lg = bsl[g];
#pragma unroll
    for (int d = 0; d < DD; ++d) lg += xrow[sub][d] * ws[d * GG + g];
    lg *= invt;
    red[sub][g] = lg; __syncthreads();
    for (int s = 32; s > 0; s >>= 1) {
        if (g < s) red[sub][g] = fmaxf(red[sub][g], red[sub][g + s]);
        __syncthreads();
    }
    float mx = red[sub][0]; __syncthreads();
    float e = __expf(lg - mx);
    red[sub][g] = e; __syncthreads();
    for (int s = 32; s > 0; s >>= 1) {
        if (g < s) red[sub][g] += red[sub][g + s];
        __syncthreads();
    }
    sw[tok * GG + g] = f2bf(e / red[sub][0]);
}

// ---------------------------------------------------------------------------
// slice_token = einsum('bhnc,bhng->bhgc') and slice_norm = sum_n sw.
// grid: (N/2048, B*H); block 256. LDS-tiled outer product, global atomics.
// Thread owns d = t&31, g in [ (t>>5)*8, +8 ).
// ---------------------------------------------------------------------------
__global__ __launch_bounds__(256) void slice_reduce(
    const __bf16* __restrict__ sw, const __bf16* __restrict__ fxm,
    float* __restrict__ acc, float* __restrict__ norm) {
    __shared__ __bf16 sws[64][GG];
    __shared__ __bf16 sfx[64][DD];
    const int bh = blockIdx.y;
    const long long swb = (long long)bh * NTOK * GG;
    const long long fxb = (long long)bh * NTOK * DD;
    const int t = threadIdx.x;
    const int d = t & 31, gi = t >> 5;
    float a[8] = {}; float nrm[8] = {};
    const int n0 = blockIdx.x * 2048;
    for (int tile = 0; tile < 2048; tile += 64) {
        __syncthreads();
        for (int i = t; i < 64 * GG; i += 256)
            sws[i >> 6][i & 63] = sw[swb + (long long)(n0 + tile + (i >> 6)) * GG + (i & 63)];
        for (int i = t; i < 64 * DD; i += 256)
            sfx[i >> 5][i & 31] = fxm[fxb + (long long)(n0 + tile + (i >> 5)) * DD + (i & 31)];
        __syncthreads();
        for (int n = 0; n < 64; ++n) {
            float fv = bf2f(sfx[n][d]);
#pragma unroll
            for (int j = 0; j < 8; ++j) {
                float sv = bf2f(sws[n][gi * 8 + j]);
                a[j] += sv * fv;
                nrm[j] += sv;
            }
        }
    }
#pragma unroll
    for (int j = 0; j < 8; ++j)
        atomicAdd(&acc[(long long)bh * (GG * DD) + (gi * 8 + j) * DD + d], a[j]);
    if (d == 0)
#pragma unroll
        for (int j = 0; j < 8; ++j)
            atomicAdd(&norm[bh * GG + gi * 8 + j], nrm[j]);
}

// ---------------------------------------------------------------------------
// Attention over G=64 slice tokens per (b,h). One 256-thread block per (b,h).
// Stores out_token transposed: outT[bh][d][g] (bf16) for de-slice B-fragments.
// ---------------------------------------------------------------------------
__global__ __launch_bounds__(256) void slice_attention(
    const float* __restrict__ acc, const float* __restrict__ norm,
    const float* __restrict__ wq, const float* __restrict__ wk,
    const float* __restrict__ wv, __bf16* __restrict__ outT) {
    __shared__ float st[GG][DD];
    __shared__ float q[GG][DD], kk[GG][DD], v[GG][DD];
    __shared__ float at[GG][GG];
    const int bh = blockIdx.x;
    const int t = threadIdx.x;
    for (int i = t; i < GG * DD; i += 256) {
        int g = i >> 5;
        st[g][i & 31] = acc[(long long)bh * (GG * DD) + i] / (norm[bh * GG + g] + 1e-5f);
    }
    __syncthreads();
    for (int i = t; i < GG * DD; i += 256) {
        int g = i >> 5, d = i & 31;
        float sq = 0, sk = 0, sv = 0;
        for (int e = 0; e < DD; ++e) {
            float s = st[g][e];
            sq += s * wq[e * DD + d];
            sk += s * wk[e * DD + d];
            sv += s * wv[e * DD + d];
        }
        q[g][d] = sq; kk[g][d] = sk; v[g][d] = sv;
    }
    __syncthreads();
    for (int i = t; i < GG * GG; i += 256) {
        int g = i >> 6, k2 = i & 63;
        float s = 0;
        for (int e = 0; e < DD; ++e) s += q[g][e] * kk[k2][e];
        at[g][k2] = s * 0.17677669529663687f;   // 32^-0.5
    }
    __syncthreads();
    if (t < GG) {
        float mx = -1e30f;
        for (int k2 = 0; k2 < GG; ++k2) mx = fmaxf(mx, at[t][k2]);
        float s = 0;
        for (int k2 = 0; k2 < GG; ++k2) { float e = __expf(at[t][k2] - mx); at[t][k2] = e; s += e; }
        float inv = 1.0f / s;
        for (int k2 = 0; k2 < GG; ++k2) at[t][k2] *= inv;
    }
    __syncthreads();
    for (int i = t; i < GG * DD; i += 256) {
        int g = i >> 5, d = i & 31;
        float s = 0;
        for (int k2 = 0; k2 < GG; ++k2) s += at[g][k2] * v[k2][d];
        outT[(long long)bh * (DD * GG) + d * GG + g] = f2bf(s);
    }
}

// ---------------------------------------------------------------------------
// Host orchestration
// ---------------------------------------------------------------------------
extern "C" void kernel_launch(void* const* d_in, const int* in_sizes, int n_in,
                              void* d_out, int out_size, void* d_ws, size_t ws_size,
                              hipStream_t stream) {
    (void)in_sizes; (void)n_in; (void)out_size; (void)ws_size;
    const float* fx     = (const float*)d_in[0];
    const float* ln1_g  = (const float*)d_in[1];
    const float* ln1_b  = (const float*)d_in[2];
    const float* w_fx   = (const float*)d_in[3];
    const float* b_fx   = (const float*)d_in[4];
    const float* w_x    = (const float*)d_in[5];
    const float* b_x    = (const float*)d_in[6];
    const float* w_sl   = (const float*)d_in[7];
    const float* b_sl   = (const float*)d_in[8];
    const float* temp   = (const float*)d_in[9];
    const float* w_q    = (const float*)d_in[10];
    const float* w_k    = (const float*)d_in[11];
    const float* w_v    = (const float*)d_in[12];
    const float* w_o    = (const float*)d_in[13];
    const float* b_o    = (const float*)d_in[14];
    const float* ln2_g  = (const float*)d_in[15];
    const float* ln2_b  = (const float*)d_in[16];
    const float* w1     = (const float*)d_in[17];
    const float* b1     = (const float*)d_in[18];
    const float* w2     = (const float*)d_in[19];
    const float* b2     = (const float*)d_in[20];
    float* out = (float*)d_out;

    char* ws = (char*)d_ws;
    __bf16* hB   = (__bf16*)(ws);                       //  64 MB: LN(h) / LN2(h2)
    __bf16* fxmB = (__bf16*)(ws + (64ll  << 20));       //  64 MB: fx_mid (b,h,n,d) / MLP g1
    __bf16* xq   = (__bf16*)(ws + (128ll << 20));       //  64 MB: x_mid (b,h,n,d) / out_x
    __bf16* swB  = (__bf16*)(ws + (192ll << 20));       // 128 MB: sw (b,h,n,g)
    char*   misc = ws + (320ll << 20);
    __bf16* wfxT = (__bf16*)(misc);                     // 128 KB
    __bf16* wxT  = (__bf16*)(misc + (128 << 10));
    __bf16* woT  = (__bf16*)(misc + (256 << 10));
    __bf16* w1T  = (__bf16*)(misc + (384 << 10));       // 512 KB [1024][256]
    __bf16* w2T  = (__bf16*)(misc + (896 << 10));       // 512 KB [256][1024]
    float*  accS = (float*)(misc + (1536 << 10));       // 256 KB [B,H,G,D]
    float*  nrmS = (float*)(misc + (1792 << 10));       //   8 KB [B,H,G]
    __bf16* otT  = (__bf16*)(misc + (1856 << 10));      // 128 KB [B,H,D,G]

    // 1. weights -> bf16 transposed [ncols][K]
    convert_transpose<<<(256 * 256 + 255) / 256, 256, 0, stream>>>(w_fx, wfxT, 256, 256);
    convert_transpose<<<(256 * 256 + 255) / 256, 256, 0, stream>>>(w_x, wxT, 256, 256);
    convert_transpose<<<(256 * 256 + 255) / 256, 256, 0, stream>>>(w_o, woT, 256, 256);
    convert_transpose<<<(256 * 1024 + 255) / 256, 256, 0, stream>>>(w1, w1T, 256, 1024);
    convert_transpose<<<(1024 * 256 + 255) / 256, 256, 0, stream>>>(w2, w2T, 1024, 256);

    // 2. LN1
    layernorm_to_bf16<<<MROWS, 256, 0, stream>>>(fx, ln1_g, ln1_b, hB);

    // 3. projections -> head-major bf16
    dim3 gproj(256 / 64, MROWS / 128, 1);
    gemm_wmma_bf16<EPI_HEADMAJOR, 4, 256><<<gproj, 256, 0, stream>>>(
        hB, 256, wfxT, 256, b_fx, fxmB, nullptr, 0, 0, 0, 0, 0);
    gemm_wmma_bf16<EPI_HEADMAJOR, 4, 256><<<gproj, 256, 0, stream>>>(
        hB, 256, wxT, 256, b_x, xq, nullptr, 0, 0, 0, 0, 0);

    // 4. slice softmax
    slice_softmax<<<(BB * HH * NTOK) / 4, 256, 0, stream>>>(xq, w_sl, b_sl, temp, swB);

    // 5. slice token reduction
    hipMemsetAsync(accS, 0, (size_t)BB * HH * GG * DD * 4, stream);
    hipMemsetAsync(nrmS, 0, (size_t)BB * HH * GG * 4, stream);
    slice_reduce<<<dim3(NTOK / 2048, BB * HH), 256, 0, stream>>>(swB, fxmB, accS, nrmS);

    // 6. attention over slice tokens
    slice_attention<<<BB * HH, 256, 0, stream>>>(accS, nrmS, w_q, w_k, w_v, otT);

    // 7. de-slice: per (b,h) GEMM [N,64] x [64,32] -> out_x (b,n,h,d) bf16 in xq
    gemm_wmma_bf16<EPI_PLAIN, 2, 64><<<dim3(1, NTOK / 128, BB * HH), 256, 0, stream>>>(
        swB, 64, otT, 64, nullptr, xq, nullptr,
        (long long)NTOK * GG, (long long)DD * GG,
        (long long)NTOK * CC, (long long)DD, 256);

    // 8. out_x @ w_o + b_o + fx  -> fx2 (f32, d_out)
    gemm_wmma_bf16<EPI_RESID, 4, 256><<<dim3(4, MROWS / 128), 256, 0, stream>>>(
        xq, 256, woT, 256, b_o, out, fx, 0, 0, 0, 0, 256);

    // 9. LN2 on fx2
    layernorm_to_bf16<<<MROWS, 256, 0, stream>>>(out, ln2_g, ln2_b, hB);

    // 10. MLP: HID=1024 split into 4 chunks of 256, accumulated into d_out
    for (int c = 0; c < 4; ++c) {
        gemm_wmma_bf16<EPI_GELU, 4, 256><<<dim3(4, MROWS / 128), 256, 0, stream>>>(
            hB, 256, w1T + (long long)c * 256 * 256, 256, b1 + c * 256,
            fxmB, nullptr, 0, 0, 0, 0, 256);
        gemm_wmma_bf16<EPI_ACC, 4, 256><<<dim3(4, MROWS / 128), 256, 0, stream>>>(
            fxmB, 256, w2T + c * 256, 1024, (c == 0 ? b2 : nullptr),
            out, nullptr, 0, 0, 0, 0, 256);
    }
}
